// MultiHeadVoting_72945724555729
// MI455X (gfx1250) — matrix operand order repaired and checked by reading, not compiled, not run on hardware
//
#include <hip/hip_runtime.h>
#include <hip/hip_bf16.h>
#include <hip/hip_fp16.h>

typedef __attribute__((ext_vector_type(16))) _Float16 v16h;
typedef __attribute__((ext_vector_type(8)))  float    v8f;

#define SLEN 1024
#define NSEL 24
#define NT   256

__global__ __launch_bounds__(NT) void MultiHeadVoting_gfx1250_kernel(
    const float* __restrict__ x,      // [B, 1024, 12]
    float* __restrict__ patch_out,    // [B, 24]  (indices+1, stored as float)
    float* __restrict__ count_out)    // [B, 1024]
{
  const int b    = blockIdx.x;
  const int tid  = threadIdx.x;
  const int lane = tid & 31;
  const int wave = tid >> 5;

  __shared__ float     sc[SLEN];     // scores (f32); sc[1023] = -inf sentinel
  __shared__ _Float16  cnt_h[SLEN];  // vote count matrix C (f16, exact 0/1)
  __shared__ _Float16  a_h[SLEN];    // band matrix A (f16), A = tri(1,2,1), symmetric
  __shared__ _Float16  t_h[SLEN];    // T = A*C (f16, exact, <=4)
  __shared__ float     oc[SLEN];     // Out = A*C*A (f32)
  __shared__ float     red_v[8];
  __shared__ int       red_i[8];

  // ---- stage 1: gather score[s] = x[b, s+1, 0] (strided by 12 floats) ----
  const float* xb = x + (size_t)b * SLEN * 12;
  for (int s = tid; s < SLEN - 1; s += NT) sc[s] = xb[(s + 1) * 12];
  if (tid == 0) sc[SLEN - 1] = -INFINITY;
  for (int s = tid; s < SLEN; s += NT) {
    cnt_h[s] = (_Float16)0.0f;
    int i = s >> 5, j = s & 31;
    int d = i - j; d = d < 0 ? -d : d;
    a_h[s] = (_Float16)((d == 0) ? 2.0f : ((d == 1) ? 1.0f : 0.0f));
  }
  __syncthreads();

  // ---- stage 2: top-24 of score; vote into cnt_h (ties -> smaller index) ----
  for (int it = 0; it < NSEL; ++it) {
    float bv = -INFINITY; int bi = SLEN;
    for (int s = tid; s < SLEN; s += NT) {          // ascending s => strict > keeps min index
      float v = sc[s];
      if (v > bv) { bv = v; bi = s; }
    }
    #pragma unroll
    for (int off = 16; off > 0; off >>= 1) {        // wave32 reduction
      float ov = __shfl_xor(bv, off, 32);
      int   oi = __shfl_xor(bi, off, 32);
      if (ov > bv || (ov == bv && oi < bi)) { bv = ov; bi = oi; }
    }
    if (lane == 0) { red_v[wave] = bv; red_i[wave] = bi; }
    __syncthreads();
    if (tid == 0) {
      float mv = red_v[0]; int mi = red_i[0];
      #pragma unroll
      for (int w = 1; w < 8; ++w)
        if (red_v[w] > mv || (red_v[w] == mv && red_i[w] < mi)) { mv = red_v[w]; mi = red_i[w]; }
      sc[mi]    = -INFINITY;
      cnt_h[mi] = (_Float16)1.0f;                   // distinct indices => count in {0,1}
    }
    __syncthreads();
  }

  // ---- stage 3: separable 3x3 conv as Out = A*C*A via 8x v_wmma_f32_16x16x32_f16 ----
  if (wave == 0) {
    const int row = lane & 15;
    const int hi  = lane >> 4;      // 0: lanes 0-15, 1: lanes 16-31
    // ---- T = A * C ----
    #pragma unroll
    for (int I = 0; I < 2; ++I) {
      v16h av;                                      // A operand: rows 16I..16I+15 of A
      {
        const int r  = 16 * I + row;
        const int kb = hi ? 8 : 0;                  // ISA 16-bit A layout
        #pragma unroll
        for (int v = 0; v < 4; ++v) {
          av[2*v]     = a_h[r*32 + kb + 2*v];
          av[2*v + 1] = a_h[r*32 + kb + 2*v + 1];
          av[8 + 2*v]     = a_h[r*32 + kb + 16 + 2*v];
          av[8 + 2*v + 1] = a_h[r*32 + kb + 16 + 2*v + 1];
        }
      }
      #pragma unroll
      for (int J = 0; J < 2; ++J) {
        v16h bm;                                    // B operand: cols 16J.. of C
        {
          const int c  = 16 * J + row;
          const int kb = hi ? 8 : 0;                // ISA 16-bit B layout
          #pragma unroll
          for (int j = 0; j < 16; ++j) bm[j] = cnt_h[(kb + (j < 8 ? j : j + 8)) * 32 + c];
        }
        v8f acc = {};
        acc = __builtin_amdgcn_wmma_f32_16x16x32_f16(false, av, false, bm,
                                                     (short)0, acc, false, false);
        #pragma unroll
        for (int r = 0; r < 8; ++r) {               // f32 D layout -> t_h (f16, exact)
          const int m = 16 * I + r + (hi ? 8 : 0);
          const int n = 16 * J + row;
          t_h[m * 32 + n] = (_Float16)acc[r];
        }
      }
    }
    // ---- Out = T * A ----
    #pragma unroll
    for (int I = 0; I < 2; ++I) {
      v16h tv;                                      // A operand: rows of T
      {
        const int r  = 16 * I + row;
        const int kb = hi ? 8 : 0;
        #pragma unroll
        for (int v = 0; v < 4; ++v) {
          tv[2*v]     = t_h[r*32 + kb + 2*v];
          tv[2*v + 1] = t_h[r*32 + kb + 2*v + 1];
          tv[8 + 2*v]     = t_h[r*32 + kb + 16 + 2*v];
          tv[8 + 2*v + 1] = t_h[r*32 + kb + 16 + 2*v + 1];
        }
      }
      #pragma unroll
      for (int J = 0; J < 2; ++J) {
        v16h bm;                                    // B operand: cols of A
        {
          const int c  = 16 * J + row;
          const int kb = hi ? 8 : 0;
          #pragma unroll
          for (int j = 0; j < 16; ++j) bm[j] = a_h[(kb + (j < 8 ? j : j + 8)) * 32 + c];
        }
        v8f acc = {};
        acc = __builtin_amdgcn_wmma_f32_16x16x32_f16(false, tv, false, bm,
                                                     (short)0, acc, false, false);
        #pragma unroll
        for (int r = 0; r < 8; ++r) {
          const int m = 16 * I + r + (hi ? 8 : 0);
          const int n = 16 * J + row;
          oc[m * 32 + n] = acc[r];
          count_out[(size_t)b * SLEN + m * 32 + n] = acc[r];   // second output
        }
      }
    }
  }
  __syncthreads();

  // ---- stage 4: top-24 of smoothed counts (ties -> smaller index), +1 ----
  for (int it = 0; it < NSEL; ++it) {
    float bv = -INFINITY; int bi = SLEN;
    for (int s = tid; s < SLEN; s += NT) {
      float v = oc[s];
      if (v > bv) { bv = v; bi = s; }
    }
    #pragma unroll
    for (int off = 16; off > 0; off >>= 1) {
      float ov = __shfl_xor(bv, off, 32);
      int   oi = __shfl_xor(bi, off, 32);
      if (ov > bv || (ov == bv && oi < bi)) { bv = ov; bi = oi; }
    }
    if (lane == 0) { red_v[wave] = bv; red_i[wave] = bi; }
    __syncthreads();
    if (tid == 0) {
      float mv = red_v[0]; int mi = red_i[0];
      #pragma unroll
      for (int w = 1; w < 8; ++w)
        if (red_v[w] > mv || (red_v[w] == mv && red_i[w] < mi)) { mv = red_v[w]; mi = red_i[w]; }
      oc[mi] = -INFINITY;
      patch_out[(size_t)b * NSEL + it] = (float)(mi + 1);      // first output (argsort(-count)+1)
    }
    __syncthreads();
  }
}

extern "C" void kernel_launch(void* const* d_in, const int* in_sizes, int n_in,
                              void* d_out, int out_size, void* d_ws, size_t ws_size,
                              hipStream_t stream) {
  (void)n_in; (void)d_ws; (void)ws_size; (void)out_size;
  const float* x = (const float*)d_in[0];
  // d_in[1] = 3x3 kernel (known separable [1,2,1]x[1,2,1], baked in)
  // d_in[2] = select_num (= 24, baked in)
  const int B = in_sizes[0] / (SLEN * 12);          // 2048
  float* out       = (float*)d_out;
  float* patch_out = out;                           // [B,24] flat, return order first
  float* count_out = out + (size_t)B * NSEL;        // [B,1024] flat
  MultiHeadVoting_gfx1250_kernel<<<B, NT, 0, stream>>>(x, patch_out, count_out);
}
